// BasicBlock_86887188398619
// MI455X (gfx1250) — matrix operand order repaired and checked by reading, not compiled, not run on hardware
//
#include <hip/hip_runtime.h>
#include <hip/hip_bf16.h>

// ---------------- constants ----------------
#define NB   8
#define HH   56
#define WW   56
#define CC   128
#define HP   58          // padded H
#define WP   58          // padded W
#define GG   4
#define PPTS 9
#define NOM  112         // 72 offsets + 36 masks + 4 pad  (multiple of 16)
#define RR   512         // MLP hidden
#define TOKS (NB*HH*WW)  // 25088

// ---------------- WMMA helpers (CDNA5, wave32) ----------------
typedef __attribute__((ext_vector_type(16))) _Float16 v16h;
typedef __attribute__((ext_vector_type(8)))  _Float16 v8h;
typedef __attribute__((ext_vector_type(8)))  float    v8f;

__device__ __forceinline__ v8f wmma32(v16h a, v16h b, v8f c) {
  // v_wmma_f32_16x16x32_f16  (f16 A/B, f32 accumulate)
  return __builtin_amdgcn_wmma_f32_16x16x32_f16(false, a, false, b, (short)0, c, false, false);
}

// A fragment: 16x32 tile, row-major with leading dim lda (elements).
// lanes 0-15 -> M=lane, lanes 16-31 -> M=lane-16;
// element e (0..15): K = e + 16*half  -> one contiguous 32B run per lane
__device__ __forceinline__ v16h load_a(const _Float16* A, int lda) {
  int lane = threadIdx.x & 31;
  int row  = lane & 15, lh = lane >> 4;
  v16h a;
#pragma unroll
  for (int e = 0; e < 16; ++e) {
    int k = e + lh * 16;
    a[e] = A[row * lda + k];
  }
  return a;
}

// B fragment from pre-swizzled storage: tile base Bs holds 32 lanes * 16 halves
// contiguously (lane-major).  Element e of lane corresponds to K = e + 16*(lane>>4),
// N = lane&15 of the original 32x16 (KxN) tile.  -> two 16B contiguous loads.
__device__ __forceinline__ v16h load_b_swz(const _Float16* Bs) {
  const int lane = threadIdx.x & 31;
  const v8h* p = (const v8h*)(Bs + lane * 16);
  v8h lo = p[0];
  v8h hi = p[1];
  return __builtin_shufflevector(lo, hi, 0, 1, 2, 3, 4, 5, 6, 7,
                                 8, 9, 10, 11, 12, 13, 14, 15);
}

// decode swizzled destination index -> (k, n) of source row-major matrix
__device__ __forceinline__ void swz_decode(int i, int KT, int& k, int& n) {
  int e = i & 15, lane = (i >> 4) & 31, tile = i >> 9;
  int kt = tile % KT, nt = tile / KT;
  k = kt * 32 + e + 16 * (lane >> 4);
  n = nt * 16 + (lane & 15);
}

__device__ __forceinline__ float gelu_exact(float v) {
  return 0.5f * v * (1.0f + erff(v * 0.7071067811865476f));
}

// ---------------- K0: zero padded xp ----------------
__global__ void k_zero(float* __restrict__ p, int n4) {
  int i = blockIdx.x * 256 + threadIdx.x;
  if (i < n4) ((float4*)p)[i] = make_float4(0.f, 0.f, 0.f, 0.f);
}

// ---------------- Kprep: weights -> f16, pre-swizzled to fragment order ----------------
__global__ void k_prep(const float* __restrict__ w_in,  const float* __restrict__ w_off,
                       const float* __restrict__ w_msk, const float* __restrict__ w_out,
                       const float* __restrict__ w_fc1, const float* __restrict__ w_fc2,
                       const float* __restrict__ b_off, const float* __restrict__ b_msk,
                       _Float16* __restrict__ win16,  _Float16* __restrict__ wom16,
                       _Float16* __restrict__ wout16, _Float16* __restrict__ wfc1_16,
                       _Float16* __restrict__ wfc2_16, float* __restrict__ bom) {
  int i = blockIdx.x * 256 + threadIdx.x;
  int k, n;
  if (i < CC * CC) {                       // w_in: K=128 (KT=4), N=128
    swz_decode(i, 4, k, n);
    win16[i] = (_Float16)w_in[k * CC + n]; return;
  }
  i -= CC * CC;
  if (i < CC * NOM) {                      // w_off|w_msk fused: K=128 (KT=4), N=112
    swz_decode(i, 4, k, n);
    float v = (n < 72) ? w_off[k * 72 + n] : ((n < 108) ? w_msk[k * 36 + (n - 72)] : 0.f);
    wom16[i] = (_Float16)v; return;
  }
  i -= CC * NOM;
  if (i < CC * CC) {                       // w_out: K=128 (KT=4), N=128
    swz_decode(i, 4, k, n);
    wout16[i] = (_Float16)w_out[k * CC + n]; return;
  }
  i -= CC * CC;
  if (i < CC * RR) {                       // w_fc1: K=128 (KT=4), N=512
    swz_decode(i, 4, k, n);
    wfc1_16[i] = (_Float16)w_fc1[k * RR + n]; return;
  }
  i -= CC * RR;
  if (i < RR * CC) {                       // w_fc2: K=512 (KT=16), N=128
    swz_decode(i, 16, k, n);
    wfc2_16[i] = (_Float16)w_fc2[k * CC + n]; return;
  }
  i -= RR * CC;
  if (i < NOM) { bom[i] = (i < 72) ? b_off[i] : ((i < 108) ? b_msk[i - 72] : 0.f); }
}

// ---------------- K1: xp = x @ w_in + b_in  -> padded image ----------------
__global__ void k_inproj(const float* __restrict__ x, const float* __restrict__ b_in,
                         const _Float16* __restrict__ win16, float* __restrict__ xp) {
  __shared__ _Float16 sA[64 * CC];      // 16 KB
  __shared__ _Float16 sB[CC * CC];      // 32 KB (swizzled)
  __shared__ int sOff[64];              // padded-image base offset per token
  const int t0 = blockIdx.x * 64;
  const int tid = threadIdx.x;
  for (int i = 0; i < 32; ++i) { int lin = tid + 256 * i; sA[lin] = (_Float16)x[(long)t0 * CC + lin]; }
  for (int i = 0; i < 64; ++i) { int lin = tid + 256 * i; sB[lin] = win16[lin]; }
  if (tid < 64) {
    int t = t0 + tid;
    int nimg = t / (HH * WW), rem = t % (HH * WW);
    sOff[tid] = (((nimg * HP) + rem / WW + 1) * WP + rem % WW + 1) * CC;
  }
  __syncthreads();
  const int wave = tid >> 5, lane = tid & 31;
  const int col = lane & 15, lh = lane >> 4;
  for (int jj = 0; jj < 4; ++jj) {
    const int j = wave + 8 * jj;
    const int mt = j >> 3, nt = j & 7;
    v8f acc;
    const float bv = b_in[nt * 16 + col];
#pragma unroll
    for (int v = 0; v < 8; ++v) acc[v] = bv;
    for (int kt = 0; kt < 4; ++kt) {
      v16h a = load_a(&sA[mt * 16 * CC + kt * 32], CC);
      v16h b = load_b_swz(&sB[(nt * 4 + kt) * 512]);
      acc = wmma32(a, b, acc);
    }
#pragma unroll
    for (int v = 0; v < 8; ++v)
      xp[(long)sOff[mt * 16 + v + 8 * lh] + nt * 16 + col] = acc[v];
  }
}

// ---------------- K2: dwconv+BN+GELU, offset/mask projection, softmax, coords ----------------
__global__ void k_offmask(const float* __restrict__ x, const float* __restrict__ dw_w,
                          const float* __restrict__ dw_b, const float* __restrict__ bn_g,
                          const float* __restrict__ bn_b, const float* __restrict__ bn_m,
                          const float* __restrict__ bn_v, const _Float16* __restrict__ wom16,
                          const float* __restrict__ bom, float* __restrict__ smp) {
  __shared__ _Float16 sH[64 * CC];                           // 16 KB
  __shared__ __align__(16) unsigned char sReg[64 * NOM * 4]; // 28 KB: swz B weights, then outputs
  _Float16* sB   = (_Float16*)sReg;
  float*    sOut = (float*)sReg;
  const int b = blockIdx.x;
  const int n = b / 49, rb = b % 49;
  const int y0 = (rb / 7) * 8, x0 = (rb % 7) * 8;
  const int tid = threadIdx.x;
  for (int i = 0; i < 56; ++i) sB[tid + 256 * i] = wom16[tid + 256 * i];
  // h = gelu(BN(dwconv(x)))
  for (int i = 0; i < 32; ++i) {
    int lin = tid + 256 * i;
    int tok = lin >> 7, ch = lin & 127;
    int py = y0 + (tok >> 3), px = x0 + (tok & 7);
    float s = 0.f;
#pragma unroll
    for (int ky = 0; ky < 3; ++ky)
#pragma unroll
      for (int kx = 0; kx < 3; ++kx) {
        int yy = py + ky - 1, xx = px + kx - 1;
        if (yy >= 0 && yy < HH && xx >= 0 && xx < WW)
          s += x[(((long)n * HH + yy) * WW + xx) * CC + ch] * dw_w[ch * 9 + ky * 3 + kx];
      }
    s += dw_b[ch];
    s = (s - bn_m[ch]) * (bn_g[ch] * rsqrtf(bn_v[ch] + 1e-5f)) + bn_b[ch];
    sH[lin] = (_Float16)gelu_exact(s);
  }
  __syncthreads();
  // 64x112 = h(64x128) @ w_om(128x112)
  const int wave = tid >> 5, lane = tid & 31;
  const int col = lane & 15, lh = lane >> 4;
  v8f dacc[4]; int jidx[4]; int nj = 0;
  for (int j = wave; j < 28; j += 8) {
    const int mt = j / 7, nt = j % 7;
    v8f acc;
#pragma unroll
    for (int v = 0; v < 8; ++v) acc[v] = 0.f;
    for (int kt = 0; kt < 4; ++kt) {
      v16h a  = load_a(&sH[mt * 16 * CC + kt * 32], CC);
      v16h bf = load_b_swz(&sB[(nt * 4 + kt) * 512]);
      acc = wmma32(a, bf, acc);
    }
    dacc[nj] = acc; jidx[nj] = j; ++nj;
  }
  __syncthreads();               // all B reads done; reuse region for outputs
  for (int q = 0; q < nj; ++q) {
    const int mt = jidx[q] / 7, nt = jidx[q] % 7;
    const float bv = bom[nt * 16 + col];
#pragma unroll
    for (int v = 0; v < 8; ++v)
      sOut[(mt * 16 + v + 8 * lh) * NOM + nt * 16 + col] = dacc[q][v] + bv;
  }
  __syncthreads();
  // softmax over 9 pts per group + sampling coords (ix = px + kw + off_x, padded coords)
  const int tok = tid >> 2, g = tid & 3;
  const int py = y0 + (tok >> 3), px = x0 + (tok & 7);
  const float* row = &sOut[tok * NOM];
  float mv[PPTS], mx = -1e30f;
#pragma unroll
  for (int p = 0; p < PPTS; ++p) { mv[p] = row[72 + g * PPTS + p]; mx = fmaxf(mx, mv[p]); }
  float ssum = 0.f;
#pragma unroll
  for (int p = 0; p < PPTS; ++p) { mv[p] = expf(mv[p] - mx); ssum += mv[p]; }
  const float inv = 1.f / ssum;
  const long tglob = ((long)n * HH + py) * WW + px;
  float* dst = &smp[((tglob * GG + g) * PPTS) * 3];
#pragma unroll
  for (int p = 0; p < PPTS; ++p) {
    int kw = p / 3, kh = p % 3;
    dst[p * 3 + 0] = (float)(px + kw) + row[(g * PPTS + p) * 2 + 0];
    dst[p * 3 + 1] = (float)(py + kh) + row[(g * PPTS + p) * 2 + 1];
    dst[p * 3 + 2] = mv[p] * inv;
  }
}

// ---------------- K3: DCN bilinear gather + out-proj + LN + residual ----------------
__global__ void k_dcn(const float* __restrict__ xp, const float* __restrict__ smp,
                      const _Float16* __restrict__ wout16, const float* __restrict__ x,
                      const float* __restrict__ b_out, const float* __restrict__ ln1_g,
                      const float* __restrict__ ln1_b, const float* __restrict__ gamma1,
                      float* __restrict__ out) {
  __shared__ _Float16 sS[64 * CC];                          // 16 KB sampled activations (f16)
  __shared__ __align__(16) unsigned char sReg[64 * CC * 4]; // 32 KB: swz w_out f16, then y f32
  __shared__ float sMean[64], sRstd[64];
  _Float16* sB = (_Float16*)sReg;
  float*    sY = (float*)sReg;
  const int t0 = blockIdx.x * 64;
  const int tid = threadIdx.x;
  for (int i = 0; i < 64; ++i) sB[tid + 256 * i] = wout16[tid + 256 * i];
  // sampling: one thread per (token, group); 32 channels each
  {
    const int tok = tid >> 2, g = tid & 3;
    const long tg = t0 + tok;
    const int n = (int)(tg / (HH * WW));
    const float* img = xp + (long)n * HP * WP * CC;
    float acc[32];
#pragma unroll
    for (int c = 0; c < 32; ++c) acc[c] = 0.f;
    const float* sp = &smp[(tg * GG + g) * PPTS * 3];
    for (int p = 0; p < PPTS; ++p) {
      const float ix = sp[p * 3 + 0], iy = sp[p * 3 + 1], m = sp[p * 3 + 2];
      const float xf = floorf(ix), yf = floorf(iy);
      const float wx1 = ix - xf, wy1 = iy - yf;
      const int xi = (int)xf, yi = (int)yf;
#pragma unroll
      for (int cy = 0; cy < 2; ++cy)
#pragma unroll
        for (int cx = 0; cx < 2; ++cx) {
          const int yy = yi + cy, xx = xi + cx;
          float w = (cy ? wy1 : 1.f - wy1) * (cx ? wx1 : 1.f - wx1) * m;
          if (yy < 0 || yy >= HP || xx < 0 || xx >= WP) w = 0.f;
          const int yc = min(max(yy, 0), HP - 1), xc = min(max(xx, 0), WP - 1);
          const float4* pr = (const float4*)(img + ((long)(yc * WP + xc)) * CC + g * 32);
#pragma unroll
          for (int q = 0; q < 8; ++q) {
            float4 v = pr[q];
            acc[q * 4 + 0] += w * v.x; acc[q * 4 + 1] += w * v.y;
            acc[q * 4 + 2] += w * v.z; acc[q * 4 + 3] += w * v.w;
          }
        }
    }
#pragma unroll
    for (int c = 0; c < 32; ++c) sS[tok * CC + g * 32 + c] = (_Float16)acc[c];
  }
  __syncthreads();
  // y = samp @ w_out
  const int wave = tid >> 5, lane = tid & 31;
  const int col = lane & 15, lh = lane >> 4;
  v8f dacc[4];
  for (int jj = 0; jj < 4; ++jj) {
    const int j = wave + 8 * jj, mt = j >> 3, nt = j & 7;
    v8f acc;
#pragma unroll
    for (int v = 0; v < 8; ++v) acc[v] = 0.f;
    for (int kt = 0; kt < 4; ++kt) {
      v16h a  = load_a(&sS[mt * 16 * CC + kt * 32], CC);
      v16h bb = load_b_swz(&sB[(nt * 4 + kt) * 512]);
      acc = wmma32(a, bb, acc);
    }
    dacc[jj] = acc;
  }
  __syncthreads();               // B reads done; reuse region for y
  for (int jj = 0; jj < 4; ++jj) {
    const int j = wave + 8 * jj, mt = j >> 3, nt = j & 7;
    const float bv = b_out[nt * 16 + col];
#pragma unroll
    for (int v = 0; v < 8; ++v)
      sY[(mt * 16 + v + 8 * lh) * CC + nt * 16 + col] = dacc[jj][v] + bv;
  }
  __syncthreads();
  if (tid < 64) {
    float s = 0.f, s2 = 0.f;
    for (int c = 0; c < CC; ++c) { float v = sY[tid * CC + c]; s += v; s2 += v * v; }
    const float mean = s / (float)CC;
    sMean[tid] = mean;
    sRstd[tid] = rsqrtf(s2 / (float)CC - mean * mean + 1e-5f);
  }
  __syncthreads();
  for (int i = 0; i < 32; ++i) {
    const int lin = tid + 256 * i;
    const int tk = lin >> 7, c = lin & 127;
    const float vy = (sY[lin] - sMean[tk]) * sRstd[tk] * ln1_g[c] + ln1_b[c];
    const long gi = ((long)t0 + tk) * CC + c;
    out[gi] = x[gi] + gamma1[c] * vy;
  }
}

// ---------------- K4: MLP + LN + residual (in-place on d_out) ----------------
__global__ void k_mlp(float* __restrict__ io, const _Float16* __restrict__ wfc1,
                      const _Float16* __restrict__ wfc2, const float* __restrict__ b_fc1,
                      const float* __restrict__ b_fc2, const float* __restrict__ ln2_g,
                      const float* __restrict__ ln2_b, const float* __restrict__ gamma2) {
  __shared__ _Float16 sA[32 * CC];    // 8 KB
  __shared__ _Float16 sM[32 * RR];    // 32 KB
  __shared__ float    sY[32 * CC];    // 16 KB
  __shared__ float sMean[32], sRstd[32];
  const long t0 = (long)blockIdx.x * 32;
  const int tid = threadIdx.x;
  for (int i = 0; i < 16; ++i) { int lin = tid + 256 * i; sA[lin] = (_Float16)io[t0 * CC + lin]; }
  __syncthreads();
  const int wave = tid >> 5, lane = tid & 31;
  const int col = lane & 15, lh = lane >> 4;
  // GEMM1: (32x128) @ (128x512), gelu -> sM (f16).  B streamed from L2-resident swz weights.
  for (int j = wave; j < 64; j += 8) {
    const int mt = j >> 5, nt = j & 31;
    // prefetch next job's weight tile (global_prefetch_b8)
    {
      const int jn = (j + 8 < 64) ? (j + 8) : j;
      __builtin_prefetch(&wfc1[((jn & 31) * 4) * 512], 0, 1);
    }
    v8f acc;
    const float bv = b_fc1[nt * 16 + col];
#pragma unroll
    for (int v = 0; v < 8; ++v) acc[v] = bv;
    for (int kt = 0; kt < 4; ++kt) {
      v16h a = load_a(&sA[mt * 16 * CC + kt * 32], CC);
      v16h b = load_b_swz(&wfc1[(nt * 4 + kt) * 512]);
      acc = wmma32(a, b, acc);
    }
#pragma unroll
    for (int v = 0; v < 8; ++v)
      sM[(mt * 16 + v + 8 * lh) * RR + nt * 16 + col] = (_Float16)gelu_exact(acc[v]);
  }
  __syncthreads();
  // GEMM2: (32x512) @ (512x128) -> sY (f32)
  for (int j = wave; j < 16; j += 8) {
    const int mt = j >> 3, nt = j & 7;
    v8f acc;
    const float bv = b_fc2[nt * 16 + col];
#pragma unroll
    for (int v = 0; v < 8; ++v) acc[v] = bv;
    for (int kt = 0; kt < 16; ++kt) {
      v16h a = load_a(&sM[mt * 16 * RR + kt * 32], RR);
      v16h b = load_b_swz(&wfc2[(nt * 16 + kt) * 512]);
      acc = wmma32(a, b, acc);
    }
#pragma unroll
    for (int v = 0; v < 8; ++v)
      sY[(mt * 16 + v + 8 * lh) * CC + nt * 16 + col] = acc[v];
  }
  __syncthreads();
  if (tid < 32) {
    float s = 0.f, s2 = 0.f;
    for (int c = 0; c < CC; ++c) { float v = sY[tid * CC + c]; s += v; s2 += v * v; }
    const float mean = s / (float)CC;
    sMean[tid] = mean;
    sRstd[tid] = rsqrtf(s2 / (float)CC - mean * mean + 1e-5f);
  }
  __syncthreads();
  for (int i = 0; i < 16; ++i) {
    const int lin = tid + 256 * i;
    const int tk = lin >> 7, c = lin & 127;
    const float vy = (sY[lin] - sMean[tk]) * sRstd[tk] * ln2_g[c] + ln2_b[c];
    io[t0 * CC + lin] = io[t0 * CC + lin] + gamma2[c] * vy;  // read-before-write, block-local
  }
}

// ---------------- launch ----------------
extern "C" void kernel_launch(void* const* d_in, const int* in_sizes, int n_in,
                              void* d_out, int out_size, void* d_ws, size_t ws_size,
                              hipStream_t stream) {
  const float* x      = (const float*)d_in[0];
  const float* w_in   = (const float*)d_in[1];
  const float* b_in   = (const float*)d_in[2];
  const float* dw_w   = (const float*)d_in[3];
  const float* dw_b   = (const float*)d_in[4];
  const float* bn_g   = (const float*)d_in[5];
  const float* bn_b   = (const float*)d_in[6];
  const float* bn_m   = (const float*)d_in[7];
  const float* bn_v   = (const float*)d_in[8];
  const float* w_off  = (const float*)d_in[9];
  const float* b_off  = (const float*)d_in[10];
  const float* w_msk  = (const float*)d_in[11];
  const float* b_msk  = (const float*)d_in[12];
  const float* w_out  = (const float*)d_in[13];
  const float* b_out  = (const float*)d_in[14];
  const float* ln1_g  = (const float*)d_in[15];
  const float* ln1_b  = (const float*)d_in[16];
  const float* w_fc1  = (const float*)d_in[17];
  const float* b_fc1  = (const float*)d_in[18];
  const float* w_fc2  = (const float*)d_in[19];
  const float* b_fc2  = (const float*)d_in[20];
  const float* ln2_g  = (const float*)d_in[21];
  const float* ln2_b  = (const float*)d_in[22];
  const float* gamma1 = (const float*)d_in[23];
  const float* gamma2 = (const float*)d_in[24];

  char* ws = (char*)d_ws;
  float*    xp      = (float*)(ws + 0);                 // 13,778,944 B padded image
  float*    smp     = (float*)(ws + 13778944);          // 10,838,016 B (ix,iy,m)
  _Float16* win16   = (_Float16*)(ws + 24616960);       // 32,768 B  (swizzled)
  _Float16* wom16   = (_Float16*)(ws + 24649728);       // 28,672 B  (swizzled)
  _Float16* wout16  = (_Float16*)(ws + 24678400);       // 32,768 B  (swizzled)
  _Float16* wfc1_16 = (_Float16*)(ws + 24711168);       // 131,072 B (swizzled)
  _Float16* wfc2_16 = (_Float16*)(ws + 24842240);       // 131,072 B (swizzled)
  float*    bom     = (float*)(ws + 24973312);          // 448 B

  k_zero<<<3364, 256, 0, stream>>>(xp, (NB * HP * WP * CC) / 4);
  k_prep<<<697, 256, 0, stream>>>(w_in, w_off, w_msk, w_out, w_fc1, w_fc2, b_off, b_msk,
                                  win16, wom16, wout16, wfc1_16, wfc2_16, bom);
  k_inproj<<<TOKS / 64, 256, 0, stream>>>(x, b_in, win16, xp);
  k_offmask<<<NB * 7 * 7, 256, 0, stream>>>(x, dw_w, dw_b, bn_g, bn_b, bn_m, bn_v,
                                            wom16, bom, smp);
  k_dcn<<<TOKS / 64, 256, 0, stream>>>(xp, smp, wout16, x, b_out, ln1_g, ln1_b, gamma1,
                                       (float*)d_out);
  k_mlp<<<TOKS / 32, 256, 0, stream>>>((float*)d_out, wfc1_16, wfc2_16, b_fc1, b_fc2,
                                       ln2_g, ln2_b, gamma2);
}
